// DynamicQuantizedLinear_1563368096609
// MI455X (gfx1250) — matrix operand order repaired and checked by reading, not compiled, not run on hardware
//
#include <hip/hip_runtime.h>
#include <stdint.h>

#define IN_DIM   4096
#define OUT_DIM  11008
#define N_TOK    32
#define WAVES_PB 4
#define KSPLIT   4
#define CHUNK_K  1024              // K depth staged in LDS per pass
#define ROW_H    1024              // halves per LDS row (= CHUNK_K)
#define BLOCK_T  (WAVES_PB * 32)

#if defined(__has_builtin)
#if __has_builtin(__builtin_amdgcn_global_load_async_to_lds_b128) && \
    __has_builtin(__builtin_amdgcn_s_wait_asynccnt)
#define USE_ASYNC_LDS 1
#endif
#endif

typedef __attribute__((ext_vector_type(16))) _Float16 v16h;
typedef __attribute__((ext_vector_type(8)))  _Float16 v8h;
typedef __attribute__((ext_vector_type(2)))  _Float16 h2;
typedef __attribute__((ext_vector_type(8)))  float    v8f;
typedef __attribute__((ext_vector_type(4)))  uint32_t u32x4;
typedef __attribute__((ext_vector_type(4)))  int      i32x4;   // param type of async builtin

// int8 -> f16 exact dequant, 16 weights at a time, using the 1024-magic trick:
//   u = w ^ 0x80  (= w + 128, linear)
//   f16(0x6400 | u) = 1024 + u = 1152 + w   (exact; ulp at 1024 is 1.0)
//   subtract 1152 (exact in f16) -> w
// 4 xor + 8 v_perm_b32 + 8 v_pk_add_f16 per 16 weights (~1.25 VALU/byte).
__device__ __forceinline__ v16h dequant16(u32x4 w) {
  union { v16h v; uint32_t u[8]; h2 h[8]; } r;
#pragma unroll
  for (int d = 0; d < 4; ++d) {
    uint32_t u = w[d] ^ 0x80808080u;
    r.u[2 * d]     = __builtin_amdgcn_perm(0x64646464u, u, 0x05010400u); // bytes 0,1 -> halves
    r.u[2 * d + 1] = __builtin_amdgcn_perm(0x64646464u, u, 0x05030402u); // bytes 2,3 -> halves
  }
  const h2 c1152 = {(_Float16)1152.0f, (_Float16)1152.0f};
#pragma unroll
  for (int j = 0; j < 8; ++j) r.h[j] = r.h[j] - c1152;
  return r.v;
}

// A-fragment (16x32 f16 MxK) from XOR-swizzled LDS.
//   each lane : K = kLoc..kLoc+15 (kLoc already has +16 for hi half-wave)
// Swizzle: 16B block b of row r lives at block (b ^ (r & 15)) -> the 16 rows read by the
// 16 lanes map to 16 distinct 16B slots = all 64 banks, conflict-free ds_load_b128.
__device__ __forceinline__ v16h load_a_lds(const _Float16* ldsx, int row, int kLoc) {
  const int sw = row & 15;
  const int b0 = (((kLoc >> 3)) ^ sw) << 3;
  const int b1 = (((kLoc >> 3) + 1) ^ sw) << 3;
  const _Float16* base = ldsx + row * ROW_H;
  v8h lo = *(const v8h*)(base + b0);
  v8h hi = *(const v8h*)(base + b1);
  v16h a;
#pragma unroll
  for (int i = 0; i < 8; ++i) { a[i] = lo[i]; a[i + 8] = hi[i]; }
  return a;
}

template <bool SPLIT>
__global__ __launch_bounds__(BLOCK_T)
void qlinear_wmma_kernel(const _Float16* __restrict__ x,
                         const int8_t*   __restrict__ wq,
                         const _Float16* __restrict__ scale,
                         const _Float16* __restrict__ bias,
                         _Float16*       __restrict__ out,
                         float*          __restrict__ part,
                         int kChunk) {
  __shared__ __align__(16) _Float16 ldsx[N_TOK * ROW_H];   // 64 KB

  const int tid   = threadIdx.x;
  const int lane  = tid & 31;
  const int wave  = tid >> 5;
  const int nBase = (blockIdx.x * WAVES_PB + wave) << 4;   // 16 channels per wave
  const int kBase = blockIdx.y * kChunk;
  const int laneN = lane & 15;
  const int hiH   = lane >> 4;
  const int bKOff = hiH << 4;   // B frag: upper half-wave holds K=16..31
  const int aKOff = hiH << 4;   // A frag: upper half-wave holds K=16..31

  const int8_t* wrow = wq + (size_t)(nBase + laneN) * IN_DIM + kBase + bKOff;

  v8f acc0 = {};   // tokens 0..15
  v8f acc1 = {};   // tokens 16..31

  for (int kb = 0; kb < kChunk; kb += CHUNK_K) {
    // ---- stage x[0..31][kBase+kb .. +CHUNK_K) into LDS (16B segments, swizzled)
    const int SEGS = N_TOK * (CHUNK_K / 8);   // 4096 segments of 16B
    for (int s = tid; s < SEGS; s += BLOCK_T) {
      const int row = s >> 7;                 // 128 segments per row
      const int c16 = s & 127;
      const _Float16* g = x + (size_t)row * IN_DIM + kBase + kb + c16 * 8;
      _Float16*       l = ldsx + row * ROW_H + ((c16 ^ (row & 15)) << 3);
#if USE_ASYNC_LDS
      __builtin_amdgcn_global_load_async_to_lds_b128((i32x4*)g, (i32x4*)l, 0, 0);
#else
      *(v8h*)l = *(const v8h*)g;
#endif
    }
#if USE_ASYNC_LDS
    __builtin_amdgcn_s_wait_asynccnt(0);
#endif
    __syncthreads();

    const int8_t* wp = wrow + kb;
#pragma unroll 2
    for (int kk = 0; kk < CHUNK_K; kk += 32) {
      __builtin_prefetch(wp + kk + 512, 0, 0);        // speculative, immediate offset
      u32x4 wraw = *(const u32x4*)(wp + kk);          // 16 int8 K-values, one channel
      v16h  bf   = dequant16(wraw);

      const int kLoc = kk + aKOff;
      v16h a0 = load_a_lds(ldsx, laneN,      kLoc);
      v16h a1 = load_a_lds(ldsx, laneN + 16, kLoc);

      acc0 = __builtin_amdgcn_wmma_f32_16x16x32_f16(false, a0, false, bf,
                                                    (short)0, acc0, false, false);
      acc1 = __builtin_amdgcn_wmma_f32_16x16x32_f16(false, a1, false, bf,
                                                    (short)0, acc1, false, false);
    }
    __syncthreads();
  }

  const int o    = nBase + laneN;
  const int rOff = hiH << 3;   // C/D layout: lanes 16-31 hold rows 8..15 of the tile

  if (SPLIT) {
    float* p = part + ((size_t)blockIdx.y * N_TOK) * OUT_DIM + o;
#pragma unroll
    for (int r = 0; r < 8; ++r) {
      p[(size_t)(rOff + r)      * OUT_DIM] = acc0[r];
      p[(size_t)(16 + rOff + r) * OUT_DIM] = acc1[r];
    }
  } else {
    const float sc = (float)scale[o];
    const float bs = (float)bias[o];
#pragma unroll
    for (int r = 0; r < 8; ++r) {
      out[(size_t)(rOff + r)      * OUT_DIM + o] = (_Float16)(acc0[r] * sc + bs);
      out[(size_t)(16 + rOff + r) * OUT_DIM + o] = (_Float16)(acc1[r] * sc + bs);
    }
  }
}

// Sum KSPLIT partials, apply per-channel scale + bias, emit fp16.
__global__ void qlinear_reduce_kernel(const float*    __restrict__ part,
                                      const _Float16* __restrict__ scale,
                                      const _Float16* __restrict__ bias,
                                      _Float16*       __restrict__ out,
                                      int n) {
  int idx = blockIdx.x * blockDim.x + threadIdx.x;
  if (idx >= n) return;
  int o = idx % OUT_DIM;
  float s = 0.f;
#pragma unroll
  for (int k = 0; k < KSPLIT; ++k)
    s += part[(size_t)k * N_TOK * OUT_DIM + idx];
  out[idx] = (_Float16)(s * (float)scale[o] + (float)bias[o]);
}

extern "C" void kernel_launch(void* const* d_in, const int* in_sizes, int n_in,
                              void* d_out, int out_size, void* d_ws, size_t ws_size,
                              hipStream_t stream) {
  (void)in_sizes; (void)n_in; (void)out_size;
  const _Float16* x     = (const _Float16*)d_in[0];
  const int8_t*   wq    = (const int8_t*)  d_in[1];
  const _Float16* scale = (const _Float16*)d_in[2];
  const _Float16* bias  = (const _Float16*)d_in[3];
  _Float16*       out   = (_Float16*)d_out;

  const dim3 block(BLOCK_T);
  const size_t need = (size_t)KSPLIT * N_TOK * OUT_DIM * sizeof(float);

  if (ws_size >= need) {
    // split-K: 2752 waves streaming int8 weights -> saturate 23.3 TB/s
    dim3 grid(OUT_DIM / (16 * WAVES_PB), KSPLIT);
    qlinear_wmma_kernel<true><<<grid, block, 0, stream>>>(
        x, wq, scale, bias, out, (float*)d_ws, IN_DIM / KSPLIT);
    int n = N_TOK * OUT_DIM;
    qlinear_reduce_kernel<<<(n + 255) / 256, 256, 0, stream>>>(
        (const float*)d_ws, scale, bias, out, n);
  } else {
    // fallback: single pass, full K per wave, fused epilogue
    dim3 grid(OUT_DIM / (16 * WAVES_PB), 1);
    qlinear_wmma_kernel<false><<<grid, block, 0, stream>>>(
        x, wq, scale, bias, out, nullptr, IN_DIM);
  }
}
